// BasicBlock_49727131353922
// MI455X (gfx1250) — compile-verified, matrix operand order checked by reading, not run on hardware
//
#include <hip/hip_runtime.h>
#include <hip/hip_bf16.h>
#include <stddef.h>

// ---------------------------------------------------------------------------
// Types / helpers for CDNA5 (gfx1250) wave32 WMMA
// ---------------------------------------------------------------------------
typedef __attribute__((ext_vector_type(16))) __bf16 bf16x16_t;
typedef __attribute__((ext_vector_type(8)))  float  f32x8_t;
typedef __attribute__((ext_vector_type(8)))  unsigned short u16x8_t;

__device__ __forceinline__ unsigned short f2bf(float f) {
    unsigned int u = __float_as_uint(f);
    unsigned int r = (u + 0x7FFFu + ((u >> 16) & 1u)) >> 16;  // RNE
    return (unsigned short)r;
}

union FragU { bf16x16_t v; u16x8_t h[2]; };

// Load one 16x32-bf16 A/B fragment row: per-lane two contiguous 8-elem chunks.
// Pattern per ISA 7.12.2: element i -> K = (i<8 ? i : i+8) + dOff, dOff = 8*(lane>=16).
__device__ __forceinline__ bf16x16_t load_frag(const unsigned short* rowPtr, int dOff) {
    FragU f;
    f.h[0] = *reinterpret_cast<const u16x8_t*>(rowPtr + dOff);
    f.h[1] = *reinterpret_cast<const u16x8_t*>(rowPtr + 16 + dOff);
    return f.v;
}

__device__ __forceinline__ f32x8_t wmma_bf16(bf16x16_t a, bf16x16_t b, f32x8_t c) {
    return __builtin_amdgcn_wmma_f32_16x16x32_bf16(false, a, false, b, (short)0, c, false, false);
}

#define RDIM 128
#define LDIM 384
#define DDIM 256
#define HEADS 8
#define DH 32
#define TOK (RDIM * LDIM)   // 49152

// ---------------------------------------------------------------------------
// Weight convert: f32 [K][N] -> bf16 [N][K] (transposed so GEMM B-frags are
// contiguous along K)
// ---------------------------------------------------------------------------
__global__ void convert_wt(const float* __restrict__ W, unsigned short* __restrict__ Wt,
                           int K, int N) {
    int gid = blockIdx.x * 256 + threadIdx.x;
    if (gid >= K * N) return;
    int n = gid % N, k = gid / N;
    Wt[(size_t)n * K + k] = f2bf(W[gid]);
}

// ---------------------------------------------------------------------------
// LayerNorm over D=256, one wave32 per row, output bf16
// ---------------------------------------------------------------------------
__global__ __launch_bounds__(256) void ln_rows(const float* __restrict__ X,
                                               const float* __restrict__ g,
                                               const float* __restrict__ b,
                                               unsigned short* __restrict__ Y,
                                               int rows) {
    int lane = threadIdx.x & 31, wave = threadIdx.x >> 5;
    int row = blockIdx.x * 8 + wave;
    if (row >= rows) return;
    const float* xp = X + (size_t)row * DDIM;
    float v[8], s = 0.f, s2 = 0.f;
#pragma unroll
    for (int i = 0; i < 8; ++i) { v[i] = xp[lane + 32 * i]; s += v[i]; s2 += v[i] * v[i]; }
#pragma unroll
    for (int off = 16; off; off >>= 1) { s += __shfl_xor(s, off, 32); s2 += __shfl_xor(s2, off, 32); }
    float mu = s * (1.f / DDIM);
    float var = s2 * (1.f / DDIM) - mu * mu;
    float inv = rsqrtf(var + 1e-5f);
    unsigned short* yp = Y + (size_t)row * DDIM;
#pragma unroll
    for (int i = 0; i < 8; ++i) {
        int c = lane + 32 * i;
        yp[c] = f2bf((v[i] - mu) * inv * g[c] + b[c]);
    }
}

// ---------------------------------------------------------------------------
// Generic WMMA GEMM: C[M,N] = act(A[M,K](bf16) x Bt[N,K](bf16) + bias (+resid))
// flags: 1 = relu, 2 = bf16 output, 4 = add f32 residual
// Block = 256 thr (8 waves), tile 64x64, each wave 16x32 (two 16x16 wmma accs)
// ---------------------------------------------------------------------------
__global__ __launch_bounds__(256) void gemm_bt(const unsigned short* __restrict__ A,
                                               const unsigned short* __restrict__ Bt,
                                               const float* __restrict__ bias,
                                               const float* __restrict__ resid,
                                               void* __restrict__ Cout,
                                               int M, int N, int K, int flags) {
    const int lane = threadIdx.x & 31, wave = threadIdx.x >> 5;
    const int nTiles = N >> 6;
    const int bm = blockIdx.x / nTiles, bn = blockIdx.x % nTiles;
    const int m0 = bm * 64 + (wave >> 1) * 16;
    const int n0 = bn * 64 + (wave & 1) * 32;
    const int ln16 = lane & 15;
    const int dOff = (lane >> 4) << 3;

    const unsigned short* aRow  = A  + (size_t)(m0 + ln16) * K;
    const unsigned short* bRow0 = Bt + (size_t)(n0 + ln16) * K;
    const unsigned short* bRow1 = Bt + (size_t)(n0 + 16 + ln16) * K;

    f32x8_t acc0 = {}, acc1 = {};
    for (int k = 0; k < K; k += 32) {
        bf16x16_t af = load_frag(aRow + k, dOff);
        bf16x16_t b0 = load_frag(bRow0 + k, dOff);
        bf16x16_t b1 = load_frag(bRow1 + k, dOff);
        acc0 = wmma_bf16(af, b0, acc0);
        acc1 = wmma_bf16(af, b1, acc1);
    }
    const int colA = n0 + ln16, colB = colA + 16;
    const int rBase = m0 + ((lane >> 4) << 3);
    float ba = bias ? bias[colA] : 0.f;
    float bb = bias ? bias[colB] : 0.f;
#pragma unroll
    for (int e = 0; e < 8; ++e) {
        int row = rBase + e;
        float v0 = acc0[e] + ba, v1 = acc1[e] + bb;
        if (flags & 1) { v0 = fmaxf(v0, 0.f); v1 = fmaxf(v1, 0.f); }
        size_t i0 = (size_t)row * N + colA, i1 = (size_t)row * N + colB;
        if (flags & 4) { v0 += resid[i0]; v1 += resid[i1]; }
        if (flags & 2) {
            ((unsigned short*)Cout)[i0] = f2bf(v0);
            ((unsigned short*)Cout)[i1] = f2bf(v1);
        } else {
            ((float*)Cout)[i0] = v0;
            ((float*)Cout)[i1] = v1;
        }
    }
}

// ---------------------------------------------------------------------------
// Width attention: one block per (l, h). scores(128x128)=Q K^T via WMMA
// (K-dim = DH = 32 = one wmma k-step), softmax with 16-lane xor reductions,
// probs staged per-wave in LDS, P@V via WMMA with V^T staged in LDS.
// Output rows ordered (l*R + r) to feed the wout GEMM.
// ---------------------------------------------------------------------------
__global__ __launch_bounds__(256) void width_attn(const unsigned short* __restrict__ Q,
                                                  const unsigned short* __restrict__ KV,
                                                  unsigned short* __restrict__ O) {
    __shared__ __align__(16) unsigned short vt[32][128];       // V^T : [d][r]
    __shared__ __align__(16) unsigned short probs[8][16][128]; // per-wave bf16 probs
    const int l = blockIdx.x >> 3, h = blockIdx.x & 7;
    const int lane = threadIdx.x & 31, wave = threadIdx.x >> 5;

    for (int idx = threadIdx.x; idx < 128 * 32; idx += 256) {
        int r = idx >> 5, d = idx & 31;
        vt[d][r] = KV[((size_t)(r * LDIM + l)) * 512 + 256 + h * 32 + d];
    }
    __syncthreads();

    const int m = lane & 15;
    const int dOff = (lane >> 4) << 3;
    const int half8 = (lane >> 4) << 3;

    const unsigned short* qRow = Q + ((size_t)((16 * wave + m) * LDIM + l)) * DDIM + h * 32;
    bf16x16_t af = load_frag(qRow, dOff);

    f32x8_t sc[8];
#pragma unroll
    for (int j = 0; j < 8; ++j) {
        const unsigned short* kRow = KV + ((size_t)((16 * j + m) * LDIM + l)) * 512 + h * 32;
        f32x8_t z = {};
        sc[j] = wmma_bf16(af, load_frag(kRow, dOff), z);
    }
    const float scale = 0.17677669529663687f;  // 32^-0.5
#pragma unroll
    for (int j = 0; j < 8; ++j)
#pragma unroll
        for (int e = 0; e < 8; ++e) sc[j][e] *= scale;

    // row-wise softmax: row lives in one 16-lane half-group across 8 col tiles
#pragma unroll
    for (int e = 0; e < 8; ++e) {
        float mx = sc[0][e];
#pragma unroll
        for (int j = 1; j < 8; ++j) mx = fmaxf(mx, sc[j][e]);
#pragma unroll
        for (int off = 1; off <= 8; off <<= 1) mx = fmaxf(mx, __shfl_xor(mx, off, 32));
        float sum = 0.f;
#pragma unroll
        for (int j = 0; j < 8; ++j) { float ev = __expf(sc[j][e] - mx); sc[j][e] = ev; sum += ev; }
#pragma unroll
        for (int off = 1; off <= 8; off <<= 1) sum += __shfl_xor(sum, off, 32);
        float inv = 1.f / sum;
#pragma unroll
        for (int j = 0; j < 8; ++j) sc[j][e] *= inv;
    }
    // scatter probs into per-wave LDS in row-major (A-fragment friendly)
#pragma unroll
    for (int j = 0; j < 8; ++j)
#pragma unroll
        for (int e = 0; e < 8; ++e)
            probs[wave][e + half8][16 * j + m] = f2bf(sc[j][e]);

    // P @ V  (K = 128 over keys, N = 32 = two 16-col tiles)
    f32x8_t o0 = {}, o1 = {};
    const unsigned short* pRow = &probs[wave][m][0];
    const unsigned short* v0Row = &vt[m][0];
    const unsigned short* v1Row = &vt[16 + m][0];
#pragma unroll
    for (int kk = 0; kk < 4; ++kk) {
        bf16x16_t ap = load_frag(pRow + 32 * kk, dOff);
        o0 = wmma_bf16(ap, load_frag(v0Row + 32 * kk, dOff), o0);
        o1 = wmma_bf16(ap, load_frag(v1Row + 32 * kk, dOff), o1);
    }
#pragma unroll
    for (int e = 0; e < 8; ++e) {
        int r = 16 * wave + e + half8;
        size_t base = ((size_t)(l * RDIM + r)) * DDIM + h * 32;
        O[base + m]      = f2bf(o0[e]);
        O[base + 16 + m] = f2bf(o1[e]);
    }
}

// ---------------------------------------------------------------------------
// Height dots: dots[h,i,j] = sum_{r,d} q[r,h,i,d] k[r,h,j,d]  (K = 4096)
// One wave per 16x16 tile of [384 x 384] per head.
// ---------------------------------------------------------------------------
__global__ __launch_bounds__(256) void hdots(const unsigned short* __restrict__ Q,
                                             const unsigned short* __restrict__ KV,
                                             float* __restrict__ D) {
    const int lane = threadIdx.x & 31, wave = threadIdx.x >> 5;
    int wid = blockIdx.x * 8 + wave;          // 0 .. 8*24*24-1
    int h = wid / 576;
    int rem = wid % 576;
    int it = rem / 24, jt = rem % 24;
    const int m = lane & 15;
    const int dOff = (lane >> 4) << 3;
    f32x8_t acc = {};
    for (int r = 0; r < RDIM; ++r) {
        const unsigned short* qRow = Q  + ((size_t)(r * LDIM + 16 * it + m)) * DDIM + h * 32;
        const unsigned short* kRow = KV + ((size_t)(r * LDIM + 16 * jt + m)) * 512 + h * 32;
        acc = wmma_bf16(load_frag(qRow, dOff), load_frag(kRow, dOff), acc);
    }
    const float scale = 1.f / 64.f;  // 32^-0.5 * 128^-0.5
    int half8 = (lane >> 4) << 3;
#pragma unroll
    for (int e = 0; e < 8; ++e)
        D[((size_t)h * LDIM + 16 * it + e + half8) * LDIM + 16 * jt + m] = acc[e] * scale;
}

// ---------------------------------------------------------------------------
// Pair bias: bias[h,i,j] = sum_p LN(pair[i,j,:])[p] * plw[p,h]; adds into dots
// One wave per (i,j), 128 channels = 4/lane, full-wave xor reductions.
// ---------------------------------------------------------------------------
__global__ __launch_bounds__(256) void pair_bias_add(const float* __restrict__ P,
                                                     const float* __restrict__ g,
                                                     const float* __restrict__ b,
                                                     const float* __restrict__ W,
                                                     float* __restrict__ D) {
    const int lane = threadIdx.x & 31, wave = threadIdx.x >> 5;
    size_t ij = (size_t)blockIdx.x * 8 + wave;
    int i = (int)(ij / LDIM), j = (int)(ij % LDIM);
    const float* pp = P + ij * 128;
    float v[4], s = 0.f, s2 = 0.f;
#pragma unroll
    for (int t = 0; t < 4; ++t) { v[t] = pp[lane + 32 * t]; s += v[t]; s2 += v[t] * v[t]; }
#pragma unroll
    for (int off = 16; off; off >>= 1) { s += __shfl_xor(s, off, 32); s2 += __shfl_xor(s2, off, 32); }
    float mu = s * (1.f / 128.f);
    float var = s2 * (1.f / 128.f) - mu * mu;
    float inv = rsqrtf(var + 1e-5f);
    float acc[8] = {0, 0, 0, 0, 0, 0, 0, 0};
#pragma unroll
    for (int t = 0; t < 4; ++t) {
        int p = lane + 32 * t;
        float nv = (v[t] - mu) * inv * g[p] + b[p];
#pragma unroll
        for (int hh = 0; hh < 8; ++hh) acc[hh] += nv * W[p * 8 + hh];
    }
#pragma unroll
    for (int hh = 0; hh < 8; ++hh)
#pragma unroll
        for (int off = 16; off; off >>= 1) acc[hh] += __shfl_xor(acc[hh], off, 32);
    if (lane == 0) {
#pragma unroll
        for (int hh = 0; hh < 8; ++hh) {
            size_t idx = ((size_t)hh * LDIM + i) * LDIM + j;
            D[idx] += acc[hh];
        }
    }
}

// ---------------------------------------------------------------------------
// Height softmax over [H*L, 384] -> bf16 probs
// ---------------------------------------------------------------------------
__global__ __launch_bounds__(256) void hsoftmax(const float* __restrict__ D,
                                                unsigned short* __restrict__ Pr) {
    const int lane = threadIdx.x & 31, wave = threadIdx.x >> 5;
    int row = blockIdx.x * 8 + wave;  // h*384 + i
    const float* dp = D + (size_t)row * LDIM;
    float v[12], mx = -3.4e38f;
#pragma unroll
    for (int t = 0; t < 12; ++t) { v[t] = dp[lane + 32 * t]; mx = fmaxf(mx, v[t]); }
#pragma unroll
    for (int off = 16; off; off >>= 1) mx = fmaxf(mx, __shfl_xor(mx, off, 32));
    float s = 0.f;
#pragma unroll
    for (int t = 0; t < 12; ++t) { v[t] = __expf(v[t] - mx); s += v[t]; }
#pragma unroll
    for (int off = 16; off; off >>= 1) s += __shfl_xor(s, off, 32);
    float inv = 1.f / s;
    unsigned short* op = Pr + (size_t)row * LDIM;
#pragma unroll
    for (int t = 0; t < 12; ++t) op[lane + 32 * t] = f2bf(v[t] * inv);
}

// ---------------------------------------------------------------------------
// Height attn output: ho[r,h,i,d] = sum_j attn[h,i,j] v[r,h,j,d]
// Block = (h, r, 64-row i-chunk); V^T staged in LDS; pure WMMA (K = 384)
// ---------------------------------------------------------------------------
__global__ __launch_bounds__(256) void hattn_o(const unsigned short* __restrict__ Pr,
                                               const unsigned short* __restrict__ KV,
                                               unsigned short* __restrict__ O) {
    __shared__ __align__(16) unsigned short vt[32][384];
    int bidx = blockIdx.x;           // h*768 + r*6 + ic
    int h = bidx / 768;
    int rem = bidx % 768;
    int r = rem / 6, ic = rem % 6;
    const int lane = threadIdx.x & 31, wave = threadIdx.x >> 5;

    for (int idx = threadIdx.x; idx < 384 * 32; idx += 256) {
        int j = idx >> 5, d = idx & 31;
        vt[d][j] = KV[((size_t)(r * LDIM + j)) * 512 + 256 + h * 32 + d];
    }
    __syncthreads();

    const int m = lane & 15;
    const int dOff = (lane >> 4) << 3;
    int i0 = 64 * ic + 16 * (wave >> 1);
    int dt = wave & 1;
    const unsigned short* pRow = Pr + ((size_t)(h * LDIM + i0 + m)) * LDIM;
    const unsigned short* vRow = &vt[16 * dt + m][0];
    f32x8_t acc = {};
#pragma unroll
    for (int kk = 0; kk < 12; ++kk)
        acc = wmma_bf16(load_frag(pRow + 32 * kk, dOff), load_frag(vRow + 32 * kk, dOff), acc);
    int half8 = (lane >> 4) << 3;
#pragma unroll
    for (int e = 0; e < 8; ++e) {
        int i = i0 + e + half8;
        O[((size_t)(r * LDIM + i)) * DDIM + h * 32 + 16 * dt + m] = f2bf(acc[e]);
    }
}

// ---------------------------------------------------------------------------
// x_mid = x + (w_out + h_out)/2   (w_out rows are (l*R+r), rest are (r*L+l))
// ---------------------------------------------------------------------------
__global__ void combine_mid(const float* __restrict__ X, const float* __restrict__ WO,
                            const float* __restrict__ HO, float* __restrict__ Out) {
    size_t gid = (size_t)blockIdx.x * 256 + threadIdx.x;
    int c = (int)(gid & 255);
    size_t t1 = gid >> 8;
    int r = (int)(t1 / LDIM), l = (int)(t1 % LDIM);
    size_t t2 = (size_t)l * RDIM + r;
    Out[gid] = X[gid] + 0.5f * (WO[t2 * DDIM + c] + HO[gid]);
}

// ---------------------------------------------------------------------------
// Host-side launch
// ---------------------------------------------------------------------------
extern "C" void kernel_launch(void* const* d_in, const int* in_sizes, int n_in,
                              void* d_out, int out_size, void* d_ws, size_t ws_size,
                              hipStream_t stream) {
    const float* x      = (const float*)d_in[0];
    const float* pair   = (const float*)d_in[1];
    const float* ln1_g  = (const float*)d_in[2];
    const float* ln1_b  = (const float*)d_in[3];
    const float* wq_w   = (const float*)d_in[4];
    const float* wkv_w  = (const float*)d_in[5];
    const float* wout_w = (const float*)d_in[6];
    const float* wout_b = (const float*)d_in[7];
    const float* hq_w   = (const float*)d_in[8];
    const float* hkv_w  = (const float*)d_in[9];
    const float* hout_w = (const float*)d_in[10];
    const float* hout_b = (const float*)d_in[11];
    const float* pn_g   = (const float*)d_in[12];
    const float* pn_b   = (const float*)d_in[13];
    const float* pl_w   = (const float*)d_in[14];
    const float* ln2_g  = (const float*)d_in[15];
    const float* ln2_b  = (const float*)d_in[16];
    const float* ff_w1  = (const float*)d_in[17];
    const float* ff_b1  = (const float*)d_in[18];
    const float* ff_w2  = (const float*)d_in[19];
    const float* ff_b2  = (const float*)d_in[20];
    float* out = (float*)d_out;

    char* ws = (char*)d_ws;
    size_t off = 0;
    auto alloc = [&](size_t sz) { size_t o = off; off += (sz + 255) & ~(size_t)255; return o; };
    const size_t T = (size_t)TOK;
    size_t off_XN   = alloc(T * 256 * 2);
    size_t off_Q    = alloc(T * 256 * 2);
    size_t off_KV   = alloc(T * 512 * 2);
    size_t off_WATT = alloc(T * 256 * 2);
    size_t off_WOUT = alloc(T * 256 * 4);
    size_t off_DOTS = alloc((size_t)8 * 384 * 384 * 4);
    size_t off_HPR  = alloc((size_t)8 * 384 * 384 * 2);
    size_t off_HOUT = alloc(T * 256 * 4);
    size_t off_wqt   = alloc(256 * 256 * 2);
    size_t off_wkvt  = alloc(256 * 512 * 2);
    size_t off_woutt = alloc(256 * 256 * 2);
    size_t off_hqt   = alloc(256 * 256 * 2);
    size_t off_hkvt  = alloc(256 * 512 * 2);
    size_t off_houtt = alloc(256 * 256 * 2);
    size_t off_ff1t  = alloc(256 * 1024 * 2);
    size_t off_ff2t  = alloc(1024 * 256 * 2);

    unsigned short* XN   = (unsigned short*)(ws + off_XN);
    unsigned short* Qb   = (unsigned short*)(ws + off_Q);
    unsigned short* KVb  = (unsigned short*)(ws + off_KV);
    unsigned short* WATT = (unsigned short*)(ws + off_WATT);
    float*          WOUT = (float*)(ws + off_WOUT);
    float*          DOTS = (float*)(ws + off_DOTS);
    unsigned short* HPR  = (unsigned short*)(ws + off_HPR);
    float*          HOUT = (float*)(ws + off_HOUT);
    // FF hidden reuses the (freed) Q + KV + WATT region: exactly T*1024*2 bytes
    unsigned short* FFH  = (unsigned short*)(ws + off_Q);

    unsigned short* wqt   = (unsigned short*)(ws + off_wqt);
    unsigned short* wkvt  = (unsigned short*)(ws + off_wkvt);
    unsigned short* woutt = (unsigned short*)(ws + off_woutt);
    unsigned short* hqt   = (unsigned short*)(ws + off_hqt);
    unsigned short* hkvt  = (unsigned short*)(ws + off_hkvt);
    unsigned short* houtt = (unsigned short*)(ws + off_houtt);
    unsigned short* ff1t  = (unsigned short*)(ws + off_ff1t);
    unsigned short* ff2t  = (unsigned short*)(ws + off_ff2t);

    // weights -> bf16 transposed
    convert_wt<<<(256 * 256 + 255) / 256, 256, 0, stream>>>(wq_w, wqt, 256, 256);
    convert_wt<<<(256 * 512 + 255) / 256, 256, 0, stream>>>(wkv_w, wkvt, 256, 512);
    convert_wt<<<(256 * 256 + 255) / 256, 256, 0, stream>>>(wout_w, woutt, 256, 256);
    convert_wt<<<(256 * 256 + 255) / 256, 256, 0, stream>>>(hq_w, hqt, 256, 256);
    convert_wt<<<(256 * 512 + 255) / 256, 256, 0, stream>>>(hkv_w, hkvt, 256, 512);
    convert_wt<<<(256 * 256 + 255) / 256, 256, 0, stream>>>(hout_w, houtt, 256, 256);
    convert_wt<<<(256 * 1024 + 255) / 256, 256, 0, stream>>>(ff_w1, ff1t, 256, 1024);
    convert_wt<<<(1024 * 256 + 255) / 256, 256, 0, stream>>>(ff_w2, ff2t, 1024, 256);

    // LN1 -> bf16
    ln_rows<<<TOK / 8, 256, 0, stream>>>(x, ln1_g, ln1_b, XN, TOK);

    // ---- width attention ----
    gemm_bt<<<(TOK / 64) * (256 / 64), 256, 0, stream>>>(XN, wqt, nullptr, nullptr, Qb, TOK, 256, 256, 2);
    gemm_bt<<<(TOK / 64) * (512 / 64), 256, 0, stream>>>(XN, wkvt, nullptr, nullptr, KVb, TOK, 512, 256, 2);
    width_attn<<<LDIM * HEADS, 256, 0, stream>>>(Qb, KVb, WATT);
    gemm_bt<<<(TOK / 64) * (256 / 64), 256, 0, stream>>>(WATT, woutt, wout_b, nullptr, WOUT, TOK, 256, 256, 0);

    // ---- height attention ----
    gemm_bt<<<(TOK / 64) * (256 / 64), 256, 0, stream>>>(XN, hqt, nullptr, nullptr, Qb, TOK, 256, 256, 2);
    gemm_bt<<<(TOK / 64) * (512 / 64), 256, 0, stream>>>(XN, hkvt, nullptr, nullptr, KVb, TOK, 512, 256, 2);
    hdots<<<(8 * 24 * 24) / 8, 256, 0, stream>>>(Qb, KVb, DOTS);
    pair_bias_add<<<(LDIM * LDIM) / 8, 256, 0, stream>>>(pair, pn_g, pn_b, pl_w, DOTS);
    hsoftmax<<<(8 * LDIM) / 8, 256, 0, stream>>>(DOTS, HPR);
    hattn_o<<<HEADS * RDIM * 6, 256, 0, stream>>>(HPR, KVb, WATT /* reused as HATT */);
    gemm_bt<<<(TOK / 64) * (256 / 64), 256, 0, stream>>>(WATT, houtt, hout_b, nullptr, HOUT, TOK, 256, 256, 0);

    // ---- residual combine -> d_out (= x_mid) ----
    combine_mid<<<TOK, 256, 0, stream>>>(x, WOUT, HOUT, out);

    // ---- feedforward with residual ----
    ln_rows<<<TOK / 8, 256, 0, stream>>>(out, ln2_g, ln2_b, XN, TOK);
    gemm_bt<<<(TOK / 64) * (1024 / 64), 256, 0, stream>>>(XN, ff1t, ff_b1, nullptr, FFH, TOK, 1024, 256, 1 | 2);
    gemm_bt<<<(TOK / 64) * (256 / 64), 256, 0, stream>>>(FFH, ff2t, ff_b2, out, out, TOK, 256, 1024, 4);
}
